// CompressedKVCache_43748536877556
// MI455X (gfx1250) — matrix operand order repaired and checked by reading, not compiled, not run on hardware
//
#include <hip/hip_runtime.h>
#include <hip/hip_bf16.h>

typedef __attribute__((ext_vector_type(16))) _Float16 v16h;
typedef __attribute__((ext_vector_type(8)))  _Float16 v8h;
typedef __attribute__((ext_vector_type(8)))  float    v8f;
typedef __attribute__((ext_vector_type(4)))  float    v4f;
typedef __attribute__((ext_vector_type(4)))  int      v4i;

#define B_   8
#define LQ   1024
#define LK   8192
#define D_   256
#define DC   128
#define EPSF 1e-6f

// ---------------------------------------------------------------------------
// Kernel 0: one-time conversions: W_up -> f16 (64 KB), scale/zero -> f16 (2 KB).
// Removes all per-block fp32->f16 conversion work from the decompress GEMM.
// ---------------------------------------------------------------------------
__global__ void prep_kernel(const float* __restrict__ Wup,
                            const float* __restrict__ ks,
                            const float* __restrict__ kz,
                            _Float16* __restrict__ Wh,
                            _Float16* __restrict__ sh,
                            _Float16* __restrict__ zh) {
  const int idx = blockIdx.x * 256 + threadIdx.x;
  if (idx < D_ * DC) Wh[idx] = (_Float16)Wup[idx];
  if (idx < B_ * DC) {
    sh[idx] = (_Float16)ks[idx];
    zh[idx] = (_Float16)kz[idx];
  }
}

// ---------------------------------------------------------------------------
// Kernel 1: q (fp32) -> q_f16 copy + per-row squared norm. One wave per row.
// ---------------------------------------------------------------------------
__global__ void qstats_kernel(const float* __restrict__ q,
                              _Float16* __restrict__ qf,
                              float* __restrict__ qsq) {
  const int wave = threadIdx.x >> 5;
  const int lane = threadIdx.x & 31;
  const int row  = blockIdx.x * 8 + wave;              // 0 .. B_*LQ-1
  const float* src = q + (size_t)row * D_ + lane * 8;  // 32 lanes * 8 = 256
  v4f a = *(const v4f*)src;
  v4f b = *(const v4f*)(src + 4);
  v8h h;
  float s = 0.f;
#pragma unroll
  for (int i = 0; i < 4; ++i) { h[i]     = (_Float16)a[i]; s += a[i] * a[i]; }
#pragma unroll
  for (int i = 0; i < 4; ++i) { h[4 + i] = (_Float16)b[i]; s += b[i] * b[i]; }
  *(v8h*)(qf + (size_t)row * D_ + lane * 8) = h;
#pragma unroll
  for (int m = 16; m >= 1; m >>= 1) s += __shfl_xor(s, m, 32);
  if (lane == 0) qsq[row] = s;
}

// ---------------------------------------------------------------------------
// Kernel 2: K decompression GEMM (fused dequant + ||k||^2).
//   k[l,d] = sum_c scale[c]*(kq[l,c]-zero[c]) * W_up[d,c]
// A-fragment: packed f16 dequant ((f16)kq is exact for codes 0..15).
// B-fragment: direct v16h load of pre-converted W_up (zero VALU).
// ||k||^2 reduced from the fp32 accumulators: shfl_xor + ds_add_f32.
// Block: 128 threads = 4 waves; covers 16 kv-rows x 256 d-cols.
// ---------------------------------------------------------------------------
__global__ void decompress_kernel(const int* __restrict__ kq,
                                  const _Float16* __restrict__ sh,
                                  const _Float16* __restrict__ zh,
                                  const _Float16* __restrict__ Wh,
                                  _Float16* __restrict__ kf,
                                  float* __restrict__ ksq) {
  const int b      = blockIdx.y;
  const int lbase  = blockIdx.x * 16;
  const int wave   = threadIdx.x >> 5;
  const int lane   = threadIdx.x & 31;
  const int lane16 = lane & 15;
  const int half   = lane >> 4;

  __shared__ float ssum[16];
  if (threadIdx.x < 16) ssum[threadIdx.x] = 0.f;
  __syncthreads();

  const _Float16* sch = sh + b * DC;
  const _Float16* zph = zh + b * DC;
  const int* kqrow = kq + ((size_t)(b * LK + lbase + lane16)) * DC;

  v8f acc[4] = {};
#pragma unroll
  for (int cb = 0; cb < DC; cb += 32) {
    // ---- A fragment: k_c = scale*(kq - zero) in f16 (ISA A layout) ----
    const int c0 = cb + half * 8;        // K = c0 + 0..7  -> elements 0..7
    const int c1 = cb + 16 + half * 8;   // K = c1 + 0..7  -> elements 8..15
    v4i i0 = *(const v4i*)(kqrow + c0);
    v4i i1 = *(const v4i*)(kqrow + c0 + 4);
    v4i i2 = *(const v4i*)(kqrow + c1);
    v4i i3 = *(const v4i*)(kqrow + c1 + 4);
    v8h s0 = *(const v8h*)(sch + c0);
    v8h s1 = *(const v8h*)(sch + c1);
    v8h z0 = *(const v8h*)(zph + c0);
    v8h z1 = *(const v8h*)(zph + c1);
    v16h afrag;
#pragma unroll
    for (int i = 0; i < 4; ++i) {
      afrag[i]      = s0[i]     * ((_Float16)i0[i] - z0[i]);
      afrag[4 + i]  = s0[4 + i] * ((_Float16)i1[i] - z0[4 + i]);
      afrag[8 + i]  = s1[i]     * ((_Float16)i2[i] - z1[i]);
      afrag[12 + i] = s1[4 + i] * ((_Float16)i3[i] - z1[4 + i]);
    }
    // ---- B fragments: pre-converted W_up f16, direct vector loads ----
#pragma unroll
    for (int t = 0; t < 4; ++t) {
      const int d = wave * 64 + t * 16 + lane16;
      v8h b0 = *(const v8h*)(Wh + (size_t)d * DC + cb + half * 8);
      v8h b1 = *(const v8h*)(Wh + (size_t)d * DC + cb + 16 + half * 8);
      v16h bfrag = __builtin_shufflevector(b0, b1, 0,1,2,3,4,5,6,7,8,9,10,11,12,13,14,15);
      acc[t] = __builtin_amdgcn_wmma_f32_16x16x32_f16(
          false, afrag, false, bfrag, (short)0, acc[t], false, false);
    }
  }

  // Store decompressed K in f16 (matrix-core-native for the main GEMM)
#pragma unroll
  for (int t = 0; t < 4; ++t) {
    const int dcol = wave * 64 + t * 16 + lane16;
#pragma unroll
    for (int r = 0; r < 8; ++r) {
      const int m = r + half * 8;
      kf[((size_t)(b * LK + lbase + m)) * D_ + dcol] = (_Float16)acc[t][r];
    }
  }

  // Fused ||k||^2: row m = r + half*8; reduce over this wave's 64 d-cols
  // (4 tiles x 16 lanes), then ds_add_f32 across waves.
#pragma unroll
  for (int r = 0; r < 8; ++r) {
    float rs = 0.f;
#pragma unroll
    for (int t = 0; t < 4; ++t) { float v = acc[t][r]; rs += v * v; }
#pragma unroll
    for (int m = 8; m >= 1; m >>= 1) rs += __shfl_xor(rs, m, 32);
    if (lane16 == 0) atomicAdd(&ssum[half * 8 + r], rs);
  }
  __syncthreads();
  if (threadIdx.x < 16) ksq[b * LK + lbase + threadIdx.x] = ssum[threadIdx.x];
}

// ---------------------------------------------------------------------------
// Kernel 3: main QK^T WMMA GEMM with fused hyperbolic-distance epilogue.
// Block: 256 threads = 8 waves -> 64(M) x 128(N) tile; wave owns 16x64.
// Output-bandwidth-bound: epilogue applied in-register, single write of D.
// ---------------------------------------------------------------------------
__global__ void dist_kernel(const _Float16* __restrict__ qf,
                            const _Float16* __restrict__ kf,
                            const float* __restrict__ qsq,
                            const float* __restrict__ ksq,
                            float* __restrict__ out) {
  const int b      = blockIdx.z;
  const int wave   = threadIdx.x >> 5;
  const int lane   = threadIdx.x & 31;
  const int lane16 = lane & 15;
  const int half   = lane >> 4;
  const int wm = wave & 3;                 // 4 waves along M
  const int wn = wave >> 2;                // 2 waves along N
  const int qrow0 = blockIdx.y * 64 + wm * 16;
  const int kcol0 = blockIdx.x * 128 + wn * 64;

  const _Float16* qrow  = qf + ((size_t)(b * LQ + qrow0 + lane16)) * D_;
  const _Float16* kbase = kf + ((size_t)(b * LK + kcol0 + lane16)) * D_;

  v8f acc[4] = {};
#pragma unroll
  for (int d0 = 0; d0 < D_; d0 += 32) {
    v16h afrag;
    v8h a0 = *(const v8h*)(qrow + d0 + half * 8);
    v8h a1 = *(const v8h*)(qrow + d0 + 16 + half * 8);
#pragma unroll
    for (int i = 0; i < 8; ++i) { afrag[i] = a0[i]; afrag[8 + i] = a1[i]; }
#pragma unroll
    for (int t = 0; t < 4; ++t) {
      v8h b0 = *(const v8h*)(kbase + (size_t)t * 16 * D_ + d0 + half * 8);
      v8h b1 = *(const v8h*)(kbase + (size_t)t * 16 * D_ + d0 + 16 + half * 8);
      v16h bfrag = __builtin_shufflevector(b0, b1, 0,1,2,3,4,5,6,7,8,9,10,11,12,13,14,15);
      acc[t] = __builtin_amdgcn_wmma_f32_16x16x32_f16(
          false, afrag, false, bfrag, (short)0, acc[t], false, false);
    }
  }

  // Fused epilogue: dist = arccosh(1 + 2*diff/denom), stable log1p form.
  float qs[8];
#pragma unroll
  for (int r = 0; r < 8; ++r) qs[r] = qsq[b * LQ + qrow0 + r + half * 8];

#pragma unroll
  for (int t = 0; t < 4; ++t) {
    const int kcol = kcol0 + t * 16 + lane16;
    const float ks = ksq[b * LK + kcol];
    const float knc = fminf(ks, 1.0f - EPSF);
#pragma unroll
    for (int r = 0; r < 8; ++r) {
      const float s    = acc[t][r];
      const float diff = fmaxf(qs[r] + ks - 2.0f * s, 0.0f);
      const float qnc  = fminf(qs[r], 1.0f - EPSF);
      const float den  = (1.0f - qnc) * (1.0f - knc) + EPSF;
      const float u    = 2.0f * diff / den;
      // arccosh(1+u) = log1p(u + sqrt(u*(u+2)))
      const float dist = log1pf(u + sqrtf(u * (u + 2.0f)));
      out[((size_t)(b * LQ + qrow0 + r + half * 8)) * LK + kcol] = dist;
    }
  }
}

// ---------------------------------------------------------------------------
extern "C" void kernel_launch(void* const* d_in, const int* in_sizes, int n_in,
                              void* d_out, int out_size, void* d_ws, size_t ws_size,
                              hipStream_t stream) {
  const float* q      = (const float*)d_in[0];   // (8,1024,256)
  const int*   kq     = (const int*)d_in[1];     // (8,8192,128)
  const float* kscale = (const float*)d_in[2];   // (8,1,128)
  const float* kzero  = (const float*)d_in[3];   // (8,1,128)
  const float* Wup    = (const float*)d_in[4];   // (256,128)
  float* out = (float*)d_out;                    // (8,1024,8192)

  char* ws = (char*)d_ws;
  _Float16* kf = (_Float16*)ws;                               // 32 MB
  size_t off = (size_t)B_ * LK * D_ * sizeof(_Float16);
  _Float16* qfp = (_Float16*)(ws + off);                      // 4 MB
  off += (size_t)B_ * LQ * D_ * sizeof(_Float16);
  float* ksq = (float*)(ws + off);                            // 256 KB
  off += (size_t)B_ * LK * sizeof(float);
  float* qsq = (float*)(ws + off);                            // 32 KB
  off += (size_t)B_ * LQ * sizeof(float);
  _Float16* Wh = (_Float16*)(ws + off);                       // 64 KB
  off += (size_t)D_ * DC * sizeof(_Float16);
  _Float16* sh = (_Float16*)(ws + off);                       // 2 KB
  off += (size_t)B_ * DC * sizeof(_Float16);
  _Float16* zh = (_Float16*)(ws + off);                       // 2 KB

  prep_kernel<<<(D_ * DC + 255) / 256, 256, 0, stream>>>(Wup, kscale, kzero, Wh, sh, zh);

  qstats_kernel<<<B_ * LQ / 8, 256, 0, stream>>>(q, qfp, qsq);

  dim3 dgrid(LK / 16, B_, 1);
  decompress_kernel<<<dgrid, 128, 0, stream>>>(kq, sh, zh, Wh, kf, ksq);

  dim3 ggrid(LK / 128, LQ / 64, B_);
  dist_kernel<<<ggrid, 256, 0, stream>>>(qfp, kf, qsq, ksq, out);
}
